// Decoder_transformer_layer_68942815035583
// MI455X (gfx1250) — compile-verified, matrix operand order checked by reading, not run on hardware
//
#include <hip/hip_runtime.h>
#include <hip/hip_bf16.h>

// ---------------------------------------------------------------------------
// CDNA5 (gfx1250) transformer decoder layer.
// B=2, S=2048, D=128, H=8, HD=16.  All matmuls on v_wmma_f32_16x16x32_f16.
// ---------------------------------------------------------------------------

typedef __attribute__((ext_vector_type(16))) _Float16 v16h;
typedef __attribute__((ext_vector_type(8)))  _Float16 v8h;
typedef __attribute__((ext_vector_type(8)))  float    v8f;

__device__ __forceinline__ v8f wmma_f16(v16h a, v16h b, v8f c) {
  // D = A(16x32 f16) * B(32x16 f16) + C(16x16 f32)
  return __builtin_amdgcn_wmma_f32_16x16x32_f16(
      /*neg_a=*/false, a, /*neg_b=*/false, b,
      /*c_mod=*/(short)0, c, /*reuse_a=*/false, /*reuse_b=*/false);
}

// Fragment with K-halves 0..15 real, 16..31 zero-padded (for HD=16 contractions)
__device__ __forceinline__ v16h frag_lo8(const _Float16* p) {
  v8h lo = *(const v8h*)p;
  v16h r;
#pragma unroll
  for (int i = 0; i < 8; ++i) { r[i] = lo[i]; r[i + 8] = (_Float16)0.0f; }
  return r;
}

// Full K=32 fragment from two contiguous 8xf16 (16B) segments
__device__ __forceinline__ v16h frag_full(const _Float16* p0, const _Float16* p1) {
  v8h lo = *(const v8h*)p0;
  v8h hi = *(const v8h*)p1;
  v16h r;
#pragma unroll
  for (int i = 0; i < 8; ++i) { r[i] = lo[i]; r[i + 8] = hi[i]; }
  return r;
}

// ---------------------------------------------------------------------------
// Prep kernels
// ---------------------------------------------------------------------------
__global__ void cvt_f32_to_f16(const float* __restrict__ x,
                               _Float16* __restrict__ y, int n) {
  int i = blockIdx.x * blockDim.x + threadIdx.x;
  if (i < n) y[i] = (_Float16)x[i];
}

// W (K x N, row-major f32)  ->  Wt (N x K, row-major f16)
__global__ void transpose_cvt(const float* __restrict__ W,
                              _Float16* __restrict__ Wt, int K, int N) {
  int idx = blockIdx.x * blockDim.x + threadIdx.x;
  if (idx >= K * N) return;
  int n = idx / K;
  int k = idx - n * K;
  Wt[idx] = (_Float16)W[(size_t)k * N + n];
}

// ---------------------------------------------------------------------------
// Generic WMMA GEMM:  C(16x16 tile per wave) = A(MxK f16) * Wt(NxK f16)^T
// mode 0: f32 out = acc + bias[n] + res[m*N+n]
// mode 1: f16 out = relu(acc + bias[n])             (row-major MxN)
// mode 2: f16 out in head layout   (b*H+h, s, hd)   [Q/K]
// mode 3: f16 out in head-T layout (b*H+h, hd, s)   [V]
// ---------------------------------------------------------------------------
__global__ void gemm_wmma(const _Float16* __restrict__ A,
                          const _Float16* __restrict__ Wt,
                          const float* __restrict__ bias,
                          const float* __restrict__ res,
                          void* __restrict__ outp,
                          int M, int N, int K, int mode) {
  int wid  = (blockIdx.x * blockDim.x + threadIdx.x) >> 5;
  int lane = threadIdx.x & 31;
  int ntn  = N >> 4;
  int total = (M >> 4) * ntn;
  if (wid >= total) return;
  int mt = wid / ntn, nt = wid - mt * ntn;
  int m0 = mt << 4, n0 = nt << 4;
  int lq = lane & 15, hf = lane >> 4;

  const _Float16* ap = A  + (size_t)(m0 + lq) * K + hf * 8;
  const _Float16* bp = Wt + (size_t)(n0 + lq) * K + hf * 8;

  v8f acc = {};
  for (int kc = 0; kc < K; kc += 32) {
    v16h af = frag_full(ap + kc, ap + kc + 16);
    v16h bf = frag_full(bp + kc, bp + kc + 16);
    acc = wmma_f16(af, bf, acc);
  }

  int n = n0 + lq;
  if (mode == 0) {
    float bv = bias[n];
    float* o = (float*)outp;
#pragma unroll
    for (int r = 0; r < 8; ++r) {
      int m = m0 + r + 8 * hf;
      o[(size_t)m * N + n] = acc[r] + bv + res[(size_t)m * N + n];
    }
  } else if (mode == 1) {
    float bv = bias[n];
    _Float16* o = (_Float16*)outp;
#pragma unroll
    for (int r = 0; r < 8; ++r) {
      int m = m0 + r + 8 * hf;
      float v = acc[r] + bv;
      o[(size_t)m * N + n] = (_Float16)(v > 0.0f ? v : 0.0f);
    }
  } else {
    // head layouts: m -> (b = m>>11, s = m&2047); n -> (hd = n>>3, h = n&7)
    int hd = n >> 3, hh = n & 7;
    _Float16* o = (_Float16*)outp;
#pragma unroll
    for (int r = 0; r < 8; ++r) {
      int m = m0 + r + 8 * hf;
      int b = m >> 11, s = m & 2047;
      size_t idx = (mode == 2)
          ? ((size_t)((b * 8 + hh) * 2048 + s) * 16 + hd)
          : ((size_t)((b * 8 + hh) * 16 + hd) * 2048 + s);
      o[idx] = (_Float16)acc[r];
    }
  }
}

// ---------------------------------------------------------------------------
// Flash attention, one wave per (b*H+h, 16-query tile), fully in registers.
// Q,K: (BH, S, 16) f16;  Vt: (BH, 16, S) f16;  out: (B,S,128) f16 merged.
// Computes S^T = K * Q^T, online softmax per-lane (+1 pair exchange),
// O^T = V^T * P^T, where score C-layout == P^T B-layout (no transpose).
// ---------------------------------------------------------------------------
__global__ void attention_wmma(const _Float16* __restrict__ Q,
                               const _Float16* __restrict__ Km,
                               const _Float16* __restrict__ Vt,
                               _Float16* __restrict__ outm) {
  constexpr int S = 2048;
  int wid  = (blockIdx.x * blockDim.x + threadIdx.x) >> 5;
  int lane = threadIdx.x & 31;
  int bh = wid >> 7;          // 0..15 (b*8 + h), 128 query tiles each
  int qt = wid & 127;
  int q0 = qt << 4;
  int lq = lane & 15, hf = lane >> 4;

  // Q^T fragment (B operand), contraction hd padded 16->32, reused all chunks
  v16h bq = frag_lo8(Q + ((size_t)bh * S + q0 + lq) * 16 + hf * 8);

  const _Float16* kbase = Km + (size_t)bh * S * 16;
  const _Float16* vbase = Vt + ((size_t)bh * 16 + lq) * S;

  v8f acc = {};
  float runM = -1e30f, runL = 0.0f;
  const float sc = 0.25f * 1.44269504088896f;  // 1/sqrt(16) * log2(e)

  for (int k0 = 0; k0 < S; k0 += 32) {
    v16h ak0 = frag_lo8(kbase + (size_t)(k0 + lq) * 16 + hf * 8);
    v16h ak1 = frag_lo8(kbase + (size_t)(k0 + 16 + lq) * 16 + hf * 8);
    v8f z = {};
    v8f st0 = wmma_f16(ak0, bq, z);   // keys (k0..k0+15)^T x queries
    v8f st1 = wmma_f16(ak1, bq, z);   // keys (k0+16..k0+31)^T x queries

    float mx = -1e30f;
#pragma unroll
    for (int i = 0; i < 8; ++i) {
      st0[i] *= sc; st1[i] *= sc;
      mx = fmaxf(mx, fmaxf(st0[i], st1[i]));
    }
    mx = fmaxf(mx, __shfl_xor(mx, 16, 32));   // lane pair holds other 8 keys
    float newM = fmaxf(runM, mx);
    float corr = exp2f(runM - newM);
    runM = newM;

    v16h bp;                                   // P^T B-fragment, direct from C-layout
    float ls = 0.0f;
#pragma unroll
    for (int i = 0; i < 8; ++i) {
      float p0 = exp2f(st0[i] - newM);
      float p1 = exp2f(st1[i] - newM);
      ls += p0 + p1;
      bp[i]     = (_Float16)p0;
      bp[i + 8] = (_Float16)p1;
    }
    runL = runL * corr + ls;
#pragma unroll
    for (int i = 0; i < 8; ++i) acc[i] *= corr;

    // V^T fragment: rows=hd (lane), cols=keys chunk (contiguous f16)
    v16h av = frag_full(vbase + k0 + hf * 8, vbase + k0 + 16 + hf * 8);
    acc = wmma_f16(av, bp, acc);               // O^T += V^T P^T
  }

  runL += __shfl_xor(runL, 16, 32);
  float inv = 1.0f / runL;

  int b = bh >> 3, h = bh & 7;
  // acc lane: query = q0+lq, hd = r + 8*hf ; merged feature = hd*8 + h
  _Float16* orow = outm + ((size_t)b * S + q0 + lq) * 128 + h;
#pragma unroll
  for (int r = 0; r < 8; ++r)
    orow[(size_t)(r + 8 * hf) * 8] = (_Float16)(acc[r] * inv);
}

// ---------------------------------------------------------------------------
// Eps-less layer norm quirk: (x - mean) / var, var unbiased (N-1=127).
// One block (128 threads) per row; f16 output feeds FFN GEMM.
// ---------------------------------------------------------------------------
__global__ void layernorm_kernel(const float* __restrict__ r2,
                                 _Float16* __restrict__ ln) {
  __shared__ float red[128];
  int row = blockIdx.x, tid = threadIdx.x;
  float x = r2[(size_t)row * 128 + tid];
  red[tid] = x; __syncthreads();
  for (int s = 64; s > 0; s >>= 1) {
    if (tid < s) red[tid] += red[tid + s];
    __syncthreads();
  }
  float mean = red[0] * (1.0f / 128.0f);
  __syncthreads();
  float c = x - mean;
  red[tid] = c * c; __syncthreads();
  for (int s = 64; s > 0; s >>= 1) {
    if (tid < s) red[tid] += red[tid + s];
    __syncthreads();
  }
  float var = red[0] * (1.0f / 127.0f);
  ln[(size_t)row * 128 + tid] = (_Float16)(c / var);
}

// ---------------------------------------------------------------------------
extern "C" void kernel_launch(void* const* d_in, const int* in_sizes, int n_in,
                              void* d_out, int out_size, void* d_ws, size_t ws_size,
                              hipStream_t stream) {
  (void)in_sizes; (void)n_in; (void)out_size; (void)ws_size;
  constexpr int Bc = 2, Sc = 2048, Dc = 128;
  constexpr int M  = Bc * Sc;              // 4096
  constexpr int NE = M * Dc;               // 524288 activation elems

  const float* x_tgt    = (const float*)d_in[0];
  const float* enc_out  = (const float*)d_in[1];
  const float* self_wq  = (const float*)d_in[2];
  const float* self_wk  = (const float*)d_in[3];
  const float* self_wv  = (const float*)d_in[4];
  const float* cross_wq = (const float*)d_in[5];
  const float* cross_wk = (const float*)d_in[6];
  const float* cross_wv = (const float*)d_in[7];
  const float* w1 = (const float*)d_in[8];  const float* b1 = (const float*)d_in[9];
  const float* w2 = (const float*)d_in[10]; const float* b2 = (const float*)d_in[11];
  const float* w3 = (const float*)d_in[12]; const float* b3 = (const float*)d_in[13];
  const float* w4 = (const float*)d_in[14]; const float* b4 = (const float*)d_in[15];
  float* out = (float*)d_out;

  char* ws = (char*)d_ws;
  size_t off = 0;
  auto alloc = [&](size_t bytes) -> void* {
    void* p = ws + off;
    off += (bytes + 255) & ~(size_t)255;
    return p;
  };

  _Float16* xt16   = (_Float16*)alloc(NE * 2);
  _Float16* enc16  = (_Float16*)alloc(NE * 2);
  _Float16* wt_sq  = (_Float16*)alloc(Dc * Dc * 2);
  _Float16* wt_sk  = (_Float16*)alloc(Dc * Dc * 2);
  _Float16* wt_sv  = (_Float16*)alloc(Dc * Dc * 2);
  _Float16* wt_cq  = (_Float16*)alloc(Dc * Dc * 2);
  _Float16* wt_ck  = (_Float16*)alloc(Dc * Dc * 2);
  _Float16* wt_cv  = (_Float16*)alloc(Dc * Dc * 2);
  _Float16* w1t    = (_Float16*)alloc(Dc * Dc * 2);
  _Float16* w2t    = (_Float16*)alloc(Dc * Dc * 2);
  _Float16* w3t    = (_Float16*)alloc(Dc * 4 * Dc * 2);
  _Float16* w4t    = (_Float16*)alloc(4 * Dc * Dc * 2);
  _Float16* qh     = (_Float16*)alloc(NE * 2);   // reused self then cross
  _Float16* kh     = (_Float16*)alloc(NE * 2);
  _Float16* vh     = (_Float16*)alloc(NE * 2);
  _Float16* sa16   = (_Float16*)alloc(NE * 2);
  _Float16* ca16   = (_Float16*)alloc(NE * 2);
  float*    r1     = (float*)alloc(NE * 4);
  float*    r2     = (float*)alloc(NE * 4);
  _Float16* ln16   = (_Float16*)alloc(NE * 2);
  _Float16* h16    = (_Float16*)alloc((size_t)M * 4 * Dc * 2);

  const int T = 256;
  // --- prep: activation cvt + weight transpose/cvt ---
  cvt_f32_to_f16<<<(NE + T - 1) / T, T, 0, stream>>>(x_tgt, xt16, NE);
  cvt_f32_to_f16<<<(NE + T - 1) / T, T, 0, stream>>>(enc_out, enc16, NE);
  int wdd = Dc * Dc, wdb = (wdd + T - 1) / T;
  transpose_cvt<<<wdb, T, 0, stream>>>(self_wq,  wt_sq, Dc, Dc);
  transpose_cvt<<<wdb, T, 0, stream>>>(self_wk,  wt_sk, Dc, Dc);
  transpose_cvt<<<wdb, T, 0, stream>>>(self_wv,  wt_sv, Dc, Dc);
  transpose_cvt<<<wdb, T, 0, stream>>>(cross_wq, wt_cq, Dc, Dc);
  transpose_cvt<<<wdb, T, 0, stream>>>(cross_wk, wt_ck, Dc, Dc);
  transpose_cvt<<<wdb, T, 0, stream>>>(cross_wv, wt_cv, Dc, Dc);
  transpose_cvt<<<wdb, T, 0, stream>>>(w1, w1t, Dc, Dc);
  transpose_cvt<<<wdb, T, 0, stream>>>(w2, w2t, Dc, Dc);
  transpose_cvt<<<(Dc * 4 * Dc + T - 1) / T, T, 0, stream>>>(w3, w3t, Dc, 4 * Dc);
  transpose_cvt<<<(4 * Dc * Dc + T - 1) / T, T, 0, stream>>>(w4, w4t, 4 * Dc, Dc);

  auto gemm_blocks = [](int m, int n) { return (((m >> 4) * (n >> 4)) * 32 + 255) / 256; };

  // --- self attention ---
  gemm_wmma<<<gemm_blocks(M, Dc), T, 0, stream>>>(xt16, wt_sq, nullptr, nullptr, qh, M, Dc, Dc, 2);
  gemm_wmma<<<gemm_blocks(M, Dc), T, 0, stream>>>(xt16, wt_sk, nullptr, nullptr, kh, M, Dc, Dc, 2);
  gemm_wmma<<<gemm_blocks(M, Dc), T, 0, stream>>>(xt16, wt_sv, nullptr, nullptr, vh, M, Dc, Dc, 3);
  attention_wmma<<<256, T, 0, stream>>>(qh, kh, vh, sa16);   // 2048 waves

  // --- cross attention (Q from encoder output; K/V from x_tgt: faithful quirk) ---
  gemm_wmma<<<gemm_blocks(M, Dc), T, 0, stream>>>(enc16, wt_cq, nullptr, nullptr, qh, M, Dc, Dc, 2);
  gemm_wmma<<<gemm_blocks(M, Dc), T, 0, stream>>>(xt16,  wt_ck, nullptr, nullptr, kh, M, Dc, Dc, 2);
  gemm_wmma<<<gemm_blocks(M, Dc), T, 0, stream>>>(xt16,  wt_cv, nullptr, nullptr, vh, M, Dc, Dc, 3);
  attention_wmma<<<256, T, 0, stream>>>(qh, kh, vh, ca16);

  // --- residual projections ---
  gemm_wmma<<<gemm_blocks(M, Dc), T, 0, stream>>>(sa16, w1t, b1, x_tgt, r1, M, Dc, Dc, 0);
  gemm_wmma<<<gemm_blocks(M, Dc), T, 0, stream>>>(ca16, w2t, b2, r1,    r2, M, Dc, Dc, 0);

  // --- layer norm (quirk: divide by unbiased variance) ---
  layernorm_kernel<<<M, Dc, 0, stream>>>(r2, ln16);

  // --- FFN ---
  gemm_wmma<<<gemm_blocks(M, 4 * Dc), T, 0, stream>>>(ln16, w3t, b3, nullptr, h16, M, 4 * Dc, Dc, 1);
  gemm_wmma<<<gemm_blocks(M, Dc), T, 0, stream>>>(h16, w4t, b4, r2, out, M, Dc, 4 * Dc, 0);
}
